// Decoder_4947802325513
// MI455X (gfx1250) — matrix operand order repaired and checked by reading, not compiled, not run on hardware
//
#include <hip/hip_runtime.h>
#include <math.h>

// Problem constants (fixed by the reference harness).
#define BB 128
#define NN 512
#define HH 128   // K of every GEMM in this model
#define TT 32
#define KC (HH / 32)  // k-chunks per GEMM = 4

typedef __attribute__((ext_vector_type(16))) _Float16 v16h;
typedef __attribute__((ext_vector_type(8)))  float    v8f;

// ---------------------------------------------------------------------------
// WMMA fragment helpers (wave32, 16x16x32 f16 -> f32; contiguous-k A/B order, ISA C/D layout)
// ---------------------------------------------------------------------------

// A fragment: 16(M) x 32(K). Source f32 row-major, converted to f16.
// lanes 0-15: M=lane, K={0..15}; lanes 16-31: M=lane-16, K={16..31}
// Per lane this is one contiguous 16-float run -> 4x global_load_b128.
__device__ inline v16h load_a_frag_f32(const float* __restrict__ A, int lda) {
  const int lane = threadIdx.x & 31;
  const int row  = lane & 15;
  const int kgrp = (lane >> 4) << 3;  // 0 or 8
  v16h a;
#pragma unroll
  for (int e = 0; e < 16; ++e) {
    const int k = e + (kgrp << 1);
    a[e] = (_Float16)A[(size_t)row * lda + k];
  }
  return a;
}

// C/D fragment store: 16x16 f32. VGPR r: lanes 0-15 -> M=r, lanes 16-31 -> M=8+r
__device__ inline void store_c_frag(float* __restrict__ C, int ldc, v8f c) {
  const int lane  = threadIdx.x & 31;
  const int col   = lane & 15;
  const int rbase = (lane >> 4) << 3;  // 0 or 8
#pragma unroll
  for (int r = 0; r < 8; ++r) C[(size_t)(rbase + r) * ldc + col] = c[r];
}

// ---------------------------------------------------------------------------
// Pack a weight W[N,K] (row-major, used as B[k][n] = W[n][k]) into WMMA
// B-fragment order: P[((tn*KC + kc)*32 + lane)*16 + e], where
//   n = tn*16 + (lane&15),  k = kc*32 + ((lane>>4)<<4) + e
// Each lane then fetches its whole fragment as one contiguous 32B read.
// ---------------------------------------------------------------------------
__global__ void pack_b_kernel(const float* __restrict__ W,
                              _Float16* __restrict__ P,
                              int N, int K) {
  const int idx = (int)(blockIdx.x * blockDim.x + threadIdx.x);
  if (idx >= N * K) return;
  const int e    = idx & 15;
  const int lane = (idx >> 4) & 31;
  const int rest = idx >> 9;
  const int kcn  = K >> 5;
  const int kc   = rest % kcn;
  const int tn   = rest / kcn;
  const int n = tn * 16 + (lane & 15);
  const int k = kc * 32 + ((lane >> 4) << 4) + e;
  P[idx] = (_Float16)W[(size_t)n * K + k];
}

// ---------------------------------------------------------------------------
// WMMA GEMM, K fixed = 128 (fully unrolled), 1 M-tile x 4 N-tiles per wave:
//   C[M,N] = A1[M,128] @ B1 (+ A2[M,128] @ B2) (+ bias0 + bias1)
// A operands f32 row-major (converted per fragment); B operands pre-packed.
// ---------------------------------------------------------------------------
__global__ void gemm_wmma_kernel(const float* __restrict__ A1,
                                 const v16h* __restrict__ B1p,
                                 const float* __restrict__ A2,
                                 const v16h* __restrict__ B2p,
                                 const float* __restrict__ bias0,
                                 const float* __restrict__ bias1,
                                 float* __restrict__ C,
                                 int M, int N) {
  const int wave  = (int)((blockIdx.x * blockDim.x + threadIdx.x) >> 5);
  const int nblk  = N >> 6;                // groups of 4 n-tiles
  const int total = (M >> 4) * nblk;
  if (wave >= total) return;  // wave-uniform: executing waves keep EXEC all-1s

  const int tm   = wave / nblk;
  const int tb   = wave - tm * nblk;       // n-tiles tb*4 .. tb*4+3
  const int lane = threadIdx.x & 31;

  v8f acc[4] = {{}, {}, {}, {}};

#pragma unroll
  for (int kc = 0; kc < KC; ++kc) {
    const v16h af = load_a_frag_f32(A1 + (size_t)tm * 16 * HH + kc * 32, HH);
#pragma unroll
    for (int j = 0; j < 4; ++j) {
      const v16h bf = B1p[((size_t)(tb * 4 + j) * KC + kc) * 32 + lane];
      acc[j] = __builtin_amdgcn_wmma_f32_16x16x32_f16(false, af, false, bf,
                                                      (short)0, acc[j],
                                                      false, false);
    }
    if (A2) {
      const v16h af2 = load_a_frag_f32(A2 + (size_t)tm * 16 * HH + kc * 32, HH);
#pragma unroll
      for (int j = 0; j < 4; ++j) {
        const v16h bf2 = B2p[((size_t)(tb * 4 + j) * KC + kc) * 32 + lane];
        acc[j] = __builtin_amdgcn_wmma_f32_16x16x32_f16(false, af2, false, bf2,
                                                        (short)0, acc[j],
                                                        false, false);
      }
    }
  }

#pragma unroll
  for (int j = 0; j < 4; ++j) {
    const int tn  = tb * 4 + j;
    const int col = tn * 16 + (lane & 15);
    float bs = 0.f;
    if (bias0) bs += bias0[col];
    if (bias1) bs += bias1[col];
#pragma unroll
    for (int r = 0; r < 8; ++r) acc[j][r] += bs;
    store_c_frag(C + (size_t)tm * 16 * N + (size_t)tn * 16, N, acc[j]);
  }
}

// ---------------------------------------------------------------------------
// Init: dec0 = mean_n(enc), hx = cx = 0, mask = 0
// ---------------------------------------------------------------------------
__global__ void init_kernel(const float* __restrict__ enc,
                            float* __restrict__ dec_input,
                            float* __restrict__ hx,
                            float* __restrict__ cx,
                            int* __restrict__ mask) {
  const int i = (int)(blockIdx.x * blockDim.x + threadIdx.x);
  if (i < BB * HH) {
    const int b = i >> 7;
    const int h = i & (HH - 1);
    float s = 0.f;
    for (int n = 0; n < NN; ++n) s += enc[((size_t)b * NN + n) * HH + h];
    dec_input[i] = s * (1.0f / (float)NN);
    hx[i] = 0.f;
    cx[i] = 0.f;
  }
  if (i < BB * NN) mask[i] = 0;
}

// ---------------------------------------------------------------------------
// LSTMCell elementwise (gates ordered i,f,g,o as torch)
// ---------------------------------------------------------------------------
__global__ void lstm_elem_kernel(const float* __restrict__ gates,
                                 float* __restrict__ hx,
                                 float* __restrict__ cx) {
  const int i = (int)(blockIdx.x * blockDim.x + threadIdx.x);
  if (i >= BB * HH) return;
  const int b = i >> 7;
  const int h = i & (HH - 1);
  const float* g = gates + (size_t)b * 4 * HH;
  const float ig = 1.f / (1.f + expf(-g[h]));
  const float fg = 1.f / (1.f + expf(-g[HH + h]));
  const float gg = tanhf(g[2 * HH + h]);
  const float og = 1.f / (1.f + expf(-g[3 * HH + h]));
  const float c = fg * cx[i] + ig * gg;
  cx[i] = c;
  hx[i] = og * tanhf(c);
}

// ---------------------------------------------------------------------------
// Attention scores + masked logsumexp + Gumbel-max categorical sample.
// One block (256 threads) per batch row. enc_trans streamed as float4.
// ---------------------------------------------------------------------------
__device__ inline unsigned wang_hash(unsigned s) {
  s = (s ^ 61u) ^ (s >> 16); s *= 9u; s = s ^ (s >> 4);
  s *= 0x27d4eb2du; s = s ^ (s >> 15);
  return s;
}

__global__ void score_sample_kernel(const float* __restrict__ enc,
                                    const float* __restrict__ enc_trans,
                                    const float* __restrict__ dec_trans,
                                    const float* __restrict__ v,
                                    int* __restrict__ mask,
                                    float* __restrict__ dec_input,
                                    float* __restrict__ out_tours,
                                    float* __restrict__ out_logp,
                                    int step) {
  const int b   = (int)blockIdx.x;
  const int tid = (int)threadIdx.x;
  const int NTH = 256;

  __shared__ float s_dec[HH];
  __shared__ float s_v[HH];
  __shared__ float s_scores[NN];
  __shared__ float s_red[256];
  __shared__ int   s_redi[256];
  __shared__ int   s_idx;

  if (tid < HH) {
    s_dec[tid] = dec_trans[(size_t)b * HH + tid];
    s_v[tid]   = v[tid];
  }
  __syncthreads();

  const float NEG_INF = -__builtin_inff();

  for (int n = tid; n < NN; n += NTH) {
    const float4* row4 =
        (const float4*)(enc_trans + ((size_t)b * NN + n) * HH);
    __builtin_prefetch((const void*)(row4 + NTH * (HH / 4)), 0, 0);
    float s = 0.f;
#pragma unroll 8
    for (int h4 = 0; h4 < HH / 4; ++h4) {
      const float4 r = row4[h4];
      s += tanhf(r.x + s_dec[4 * h4 + 0]) * s_v[4 * h4 + 0];
      s += tanhf(r.y + s_dec[4 * h4 + 1]) * s_v[4 * h4 + 1];
      s += tanhf(r.z + s_dec[4 * h4 + 2]) * s_v[4 * h4 + 2];
      s += tanhf(r.w + s_dec[4 * h4 + 3]) * s_v[4 * h4 + 3];
    }
    if (mask[(size_t)b * NN + n]) s = NEG_INF;
    s_scores[n] = s;
  }
  __syncthreads();

  // local max (for logsumexp) and local Gumbel-max candidate
  float lmax = NEG_INF;
  float lkey = NEG_INF;
  int   lidx = 0;
  for (int n = tid; n < NN; n += NTH) {
    const float s = s_scores[n];
    lmax = fmaxf(lmax, s);
    const unsigned hsh =
        wang_hash((((unsigned)step * (unsigned)BB + (unsigned)b) * (unsigned)NN +
                   (unsigned)n) ^ 0x9E3779B9u);
    const float u = ((float)hsh + 1.0f) * 2.3283064e-10f;  // (0,1]
    const float gum = -logf(-logf(u) + 1e-20f);
    const float key = s + gum;  // -inf stays -inf for masked entries
    if (key > lkey) { lkey = key; lidx = n; }
  }

  // tree-reduce max
  s_red[tid] = lmax;
  __syncthreads();
  for (int off = NTH >> 1; off > 0; off >>= 1) {
    if (tid < off) s_red[tid] = fmaxf(s_red[tid], s_red[tid + off]);
    __syncthreads();
  }
  const float m = s_red[0];
  __syncthreads();

  // tree-reduce argmax of gumbel keys
  s_red[tid]  = lkey;
  s_redi[tid] = lidx;
  __syncthreads();
  for (int off = NTH >> 1; off > 0; off >>= 1) {
    if (tid < off && s_red[tid + off] > s_red[tid]) {
      s_red[tid]  = s_red[tid + off];
      s_redi[tid] = s_redi[tid + off];
    }
    __syncthreads();
  }
  if (tid == 0) s_idx = s_redi[0];
  __syncthreads();
  const int idx = s_idx;
  __syncthreads();

  // tree-reduce sum(exp(s - m))
  float lsum = 0.f;
  for (int n = tid; n < NN; n += NTH) {
    lsum += expf(s_scores[n] - m);  // masked -inf contributes 0
  }
  s_red[tid] = lsum;
  __syncthreads();
  for (int off = NTH >> 1; off > 0; off >>= 1) {
    if (tid < off) s_red[tid] += s_red[tid + off];
    __syncthreads();
  }

  if (tid == 0) {
    const float logp = s_scores[idx] - m - logf(s_red[0]);
    out_tours[(size_t)b * TT + step] = (float)idx;
    out_logp[(size_t)b * TT + step] = logp;
    mask[(size_t)b * NN + idx] = 1;
  }
  // gather encoder_outputs[b, idx, :] as next decoder input
  if (tid < HH)
    dec_input[(size_t)b * HH + tid] = enc[((size_t)b * NN + idx) * HH + tid];
}

// ---------------------------------------------------------------------------
// Host launcher
// ---------------------------------------------------------------------------
static inline size_t align_up(size_t x, size_t a) { return (x + a - 1) & ~(a - 1); }

extern "C" void kernel_launch(void* const* d_in, const int* in_sizes, int n_in,
                              void* d_out, int out_size, void* d_ws, size_t ws_size,
                              hipStream_t stream) {
  (void)in_sizes; (void)n_in; (void)out_size; (void)ws_size;

  const float* enc  = (const float*)d_in[0];  // [B,N,H]
  const float* W_ih = (const float*)d_in[1];  // [4H,H]
  const float* W_hh = (const float*)d_in[2];  // [4H,H]
  const float* b_ih = (const float*)d_in[3];  // [4H]
  const float* b_hh = (const float*)d_in[4];  // [4H]
  const float* W1   = (const float*)d_in[5];  // [H,H]
  const float* W2   = (const float*)d_in[6];  // [H,H]
  const float* vv   = (const float*)d_in[7];  // [H]
  // d_in[8] = decode_steps (device scalar); shapes fixed -> TT compile-time

  float* out_tours = (float*)d_out;            // [B,T]
  float* out_logp  = out_tours + BB * TT;      // [B,T]

  // workspace carve-up
  char* p = (char*)d_ws;
  size_t off = 0;
  auto carve = [&](size_t bytes) {
    void* r = p + off;
    off = align_up(off + bytes, 256);
    return r;
  };
  _Float16* P_W1  = (_Float16*)carve((size_t)HH * HH * 2);
  _Float16* P_Wih = (_Float16*)carve((size_t)4 * HH * HH * 2);
  _Float16* P_Whh = (_Float16*)carve((size_t)4 * HH * HH * 2);
  _Float16* P_W2  = (_Float16*)carve((size_t)HH * HH * 2);
  float* enc_trans = (float*)carve((size_t)BB * NN * HH * 4);   // 32 MB
  float* gates     = (float*)carve((size_t)BB * 4 * HH * 4);
  float* dec_input = (float*)carve((size_t)BB * HH * 4);
  float* hx        = (float*)carve((size_t)BB * HH * 4);
  float* cx        = (float*)carve((size_t)BB * HH * 4);
  float* dec_trans = (float*)carve((size_t)BB * HH * 4);
  int*   mask      = (int*)carve((size_t)BB * NN * 4);

  // ---- one-time prep: pack weights into WMMA B-fragment order ----
  pack_b_kernel<<<(HH * HH + 255) / 256, 256, 0, stream>>>(W1, P_W1, HH, HH);
  pack_b_kernel<<<(4 * HH * HH + 255) / 256, 256, 0, stream>>>(W_ih, P_Wih, 4 * HH, HH);
  pack_b_kernel<<<(4 * HH * HH + 255) / 256, 256, 0, stream>>>(W_hh, P_Whh, 4 * HH, HH);
  pack_b_kernel<<<(HH * HH + 255) / 256, 256, 0, stream>>>(W2, P_W2, HH, HH);

  init_kernel<<<(BB * NN + 255) / 256, 256, 0, stream>>>(enc, dec_input, hx, cx, mask);

  // enc_trans[b,n,k] = sum_h enc[b,n,h] * W1[k,h]  -> M=B*N, N=H
  {
    const int M = BB * NN, N = HH;
    const int waves = (M / 16) * (N / 64);          // 8192
    gemm_wmma_kernel<<<waves / 8, 256, 0, stream>>>(
        enc, (const v16h*)P_W1, nullptr, nullptr, nullptr, nullptr,
        enc_trans, M, N);
  }

  // ---- decode loop ----
  for (int t = 0; t < TT; ++t) {
    // gates = dec_input @ W_ih^T + hx @ W_hh^T + b_ih + b_hh   [B, 4H]
    {
      const int M = BB, N = 4 * HH;
      const int waves = (M / 16) * (N / 64);        // 64
      gemm_wmma_kernel<<<waves / 8, 256, 0, stream>>>(
          dec_input, (const v16h*)P_Wih, hx, (const v16h*)P_Whh,
          b_ih, b_hh, gates, M, N);
    }
    lstm_elem_kernel<<<(BB * HH + 255) / 256, 256, 0, stream>>>(gates, hx, cx);
    // dec_trans = hx_new @ W2^T   [B, H]
    {
      const int M = BB, N = HH;
      const int waves = (M / 16) * (N / 64);        // 16
      gemm_wmma_kernel<<<waves / 8, 256, 0, stream>>>(
          hx, (const v16h*)P_W2, nullptr, nullptr, nullptr, nullptr,
          dec_trans, M, N);
    }
    score_sample_kernel<<<BB, 256, 0, stream>>>(
        enc, enc_trans, dec_trans, vv, mask, dec_input, out_tours, out_logp, t);
  }
}
